// MultilevelSelfAttentionBlockWithRoPE_69423851373333
// MI455X (gfx1250) — matrix operand order repaired and checked by reading, not compiled, not run on hardware
//
#include <hip/hip_runtime.h>

#define DM    1024
#define NH    16
#define HD    64
#define NTOK  6144
#define NBAT  8

typedef __bf16 v16bf __attribute__((ext_vector_type(16)));
typedef float  v8f   __attribute__((ext_vector_type(8)));

struct Frag32 { uint4 lo, hi; };   // 32 bytes = one 16-element bf16 fragment

static __device__ __forceinline__ v16bf as_v16bf(const Frag32& f) {
  return __builtin_bit_cast(v16bf, f);
}

static __device__ __forceinline__ v8f wmma_bf16(v16bf a, v16bf b, v8f c) {
  // D = A(16x32 bf16) * B(32x16 bf16) + C(16x16 f32)
  return __builtin_amdgcn_wmma_f32_16x16x32_bf16(false, a, false, b, (short)0, c, false, false);
}

static __device__ __forceinline__ unsigned short f2bf(float x) {
  unsigned int u = __builtin_bit_cast(unsigned int, x);
  unsigned int r = (u + 0x7FFFu + ((u >> 16) & 1u)) >> 16;
  return (unsigned short)r;
}

// ---------------------------------------------------------------------------
// Transpose+convert f32 weight [K x Ncols] -> bf16 [Ncols x K]
// ---------------------------------------------------------------------------
__global__ void convT_kernel(const float* __restrict__ w, unsigned short* __restrict__ wT,
                             int K, int Ncols) {
  int idx = blockIdx.x * blockDim.x + threadIdx.x;
  if (idx >= K * Ncols) return;
  int k = idx / Ncols;
  int n = idx - k * Ncols;
  wT[(size_t)n * K + k] = f2bf(w[idx]);
}

// ---------------------------------------------------------------------------
// LayerNorm: one 256-thread block per token row, output bf16
// ---------------------------------------------------------------------------
__global__ void ln_kernel(const float* __restrict__ x, const float* __restrict__ g,
                          const float* __restrict__ b, unsigned short* __restrict__ hout) {
  __shared__ float red[256];
  int row = blockIdx.x, t = threadIdx.x;
  const float* xr = x + (size_t)row * DM;
  float a[4];
#pragma unroll
  for (int j = 0; j < 4; ++j) a[j] = xr[t * 4 + j];
  red[t] = a[0] + a[1] + a[2] + a[3];
  __syncthreads();
  for (int o = 128; o > 0; o >>= 1) { if (t < o) red[t] += red[t + o]; __syncthreads(); }
  float mu = red[0] * (1.0f / DM);
  __syncthreads();
  float d0 = a[0] - mu, d1 = a[1] - mu, d2 = a[2] - mu, d3 = a[3] - mu;
  red[t] = d0 * d0 + d1 * d1 + d2 * d2 + d3 * d3;
  __syncthreads();
  for (int o = 128; o > 0; o >>= 1) { if (t < o) red[t] += red[t + o]; __syncthreads(); }
  float inv = rsqrtf(red[0] * (1.0f / DM) + 1e-5f);
  unsigned short* ho = hout + (size_t)row * DM + t * 4;
#pragma unroll
  for (int j = 0; j < 4; ++j)
    ho[j] = f2bf((a[j] - mu) * inv * g[t * 4 + j] + b[t * 4 + j]);
}

// ---------------------------------------------------------------------------
// QKV GEMM: qkv[N,3072] = h[N,1024] @ wqkvT'[1024,3072]; one wave per 16x64
// tile; epilogue scatters into per-head q/k layouts and transposed V.
// ---------------------------------------------------------------------------
__global__ void __launch_bounds__(32) gemm_qkv_kernel(
    const unsigned short* __restrict__ hbuf, const unsigned short* __restrict__ wT,
    unsigned short* __restrict__ qb, unsigned short* __restrict__ kb,
    unsigned short* __restrict__ vT) {
  const int MT = NTOK / 16;                 // 384
  int wid = blockIdx.x;
  int mt = wid % MT, nt = wid / MT;         // nt: 0..47
  int lane = threadIdx.x, half = lane >> 4, l16 = lane & 15;
  int row0 = mt * 16, col0 = nt * 64;
  v8f acc[4] = {};
  const unsigned short* arow = hbuf + (size_t)(row0 + l16) * DM;
  for (int k0 = 0; k0 < DM; k0 += 32) {
    Frag32 af;
    af.lo = *(const uint4*)(arow + k0 + half * 16);
    af.hi = *(const uint4*)(arow + k0 + 8 + half * 16);
    v16bf av = as_v16bf(af);
#pragma unroll
    for (int c = 0; c < 4; ++c) {
      const unsigned short* brow = wT + (size_t)(col0 + c * 16 + l16) * DM + k0 + half * 16;
      Frag32 bf; bf.lo = *(const uint4*)brow; bf.hi = *(const uint4*)(brow + 8);
      acc[c] = wmma_bf16(av, as_v16bf(bf), acc[c]);
    }
  }
  int which = col0 >> 10;                   // 0=q 1=k 2=v
  int head  = (col0 & 1023) >> 6;
#pragma unroll
  for (int c = 0; c < 4; ++c) {
#pragma unroll
    for (int r = 0; r < 8; ++r) {
      int row = row0 + half * 8 + r;
      int d = c * 16 + l16;
      unsigned short v = f2bf(acc[c][r]);
      if (which == 0)      qb[((size_t)head * NTOK + row) * HD + d] = v;
      else if (which == 1) kb[((size_t)head * NTOK + row) * HD + d] = v;
      else                 vT[((size_t)(head * HD + d)) * NTOK + row] = v;
    }
  }
}

// ---------------------------------------------------------------------------
// Multilevel RoPE on q,k in place (per-pair rotation is independent)
// ---------------------------------------------------------------------------
__global__ void rope_kernel(const float* __restrict__ pos, const float* __restrict__ freqs,
                            unsigned short* __restrict__ qb, unsigned short* __restrict__ kb) {
  int idx = blockIdx.x * blockDim.x + threadIdx.x;   // N * 16 heads * 32 pairs
  if (idx >= NTOK * NH * 32) return;
  int i = idx >> 9;
  int h = (idx >> 5) & 15;
  int j = idx & 31;
  int lvl = (int)pos[i * 3];
  lvl = lvl < 0 ? 0 : (lvl > 3 ? 3 : lvl);
  float y = pos[i * 3 + 1], xx = pos[i * 3 + 2];
  const float* f = freqs + ((size_t)((lvl * NH + h) * 32 + j)) * 2;
  float ang = f[0] * y + f[1] * xx;
  float c = cosf(ang), s = sinf(ang);
  size_t base = ((size_t)h * NTOK + i) * HD + 2 * j;
  {
    float a = __builtin_bit_cast(float, (unsigned int)qb[base] << 16);
    float b = __builtin_bit_cast(float, (unsigned int)qb[base + 1] << 16);
    qb[base] = f2bf(a * c - b * s);
    qb[base + 1] = f2bf(a * s + b * c);
  }
  {
    float a = __builtin_bit_cast(float, (unsigned int)kb[base] << 16);
    float b = __builtin_bit_cast(float, (unsigned int)kb[base + 1] << 16);
    kb[base] = f2bf(a * c - b * s);
    kb[base + 1] = f2bf(a * s + b * c);
  }
}

// ---------------------------------------------------------------------------
// Flash attention: one wave per (batch, head, 16-query tile), keys 32 at a time
// ---------------------------------------------------------------------------
__global__ void __launch_bounds__(32) attn_kernel(
    const int* __restrict__ boff, const unsigned short* __restrict__ qb,
    const unsigned short* __restrict__ kb, const unsigned short* __restrict__ vT,
    unsigned short* __restrict__ ob) {
  __shared__ unsigned short T[16 * 32];     // P transpose staging (1 KB)
  int bid = blockIdx.x;
  int qt = bid & 63;
  int h  = (bid >> 6) & 15;
  int b  = bid >> 10;
  int off0 = boff[b];
  int len  = boff[b + 1] - off0;
  int q0 = qt * 16;
  if (q0 >= len) return;
  int lane = threadIdx.x, half = lane >> 4, l16 = lane & 15;

  // Q A-fragments for the two 32-wide head_dim chunks
  const unsigned short* qrow = qb + ((size_t)h * NTOK + off0 + q0 + l16) * HD;
  Frag32 qa0f, qa1f;
  qa0f.lo = *(const uint4*)(qrow + half * 16);
  qa0f.hi = *(const uint4*)(qrow + 8 + half * 16);
  qa1f.lo = *(const uint4*)(qrow + 32 + half * 16);
  qa1f.hi = *(const uint4*)(qrow + 40 + half * 16);
  v16bf qa0 = as_v16bf(qa0f), qa1 = as_v16bf(qa1f);

  float M[8], L[8];
#pragma unroll
  for (int r = 0; r < 8; ++r) { M[r] = -3.0e38f; L[r] = 0.0f; }
  v8f o[4] = {};
  const float scale = 0.125f;               // 1/sqrt(64)

  for (int k0 = 0; k0 < len; k0 += 32) {
    v8f s0 = {}, s1 = {};
    {
      const unsigned short* kr = kb + ((size_t)h * NTOK + off0 + k0 + l16) * HD + half * 16;
      Frag32 b0, b1;
      b0.lo = *(const uint4*)kr;        b0.hi = *(const uint4*)(kr + 8);
      b1.lo = *(const uint4*)(kr + 32); b1.hi = *(const uint4*)(kr + 40);
      s0 = wmma_bf16(qa0, as_v16bf(b0), s0);
      s0 = wmma_bf16(qa1, as_v16bf(b1), s0);
    }
    {
      const unsigned short* kr = kb + ((size_t)h * NTOK + off0 + k0 + 16 + l16) * HD + half * 16;
      Frag32 b0, b1;
      b0.lo = *(const uint4*)kr;        b0.hi = *(const uint4*)(kr + 8);
      b1.lo = *(const uint4*)(kr + 32); b1.hi = *(const uint4*)(kr + 40);
      s1 = wmma_bf16(qa0, as_v16bf(b0), s1);
      s1 = wmma_bf16(qa1, as_v16bf(b1), s1);
    }
    bool v0 = (k0 + l16) < len;
    bool v1 = (k0 + 16 + l16) < len;
    float alpha[8], p0a[8], p1a[8];
#pragma unroll
    for (int r = 0; r < 8; ++r) {
      float a0 = v0 ? s0[r] * scale : -1.0e9f;
      float a1 = v1 ? s1[r] * scale : -1.0e9f;
      float m = fmaxf(a0, a1);
      for (int o2 = 1; o2 < 16; o2 <<= 1) m = fmaxf(m, __shfl_xor(m, o2, 32));
      float Mn = fmaxf(M[r], m);
      float al = __expf(M[r] - Mn);
      float p0 = __expf(a0 - Mn), p1 = __expf(a1 - Mn);
      float sum = p0 + p1;
      for (int o2 = 1; o2 < 16; o2 <<= 1) sum += __shfl_xor(sum, o2, 32);
      L[r] = L[r] * al + sum;
      M[r] = Mn;
      alpha[r] = al; p0a[r] = p0; p1a[r] = p1;
    }
#pragma unroll
    for (int dd = 0; dd < 4; ++dd)
#pragma unroll
      for (int r = 0; r < 8; ++r) o[dd][r] *= alpha[r];
    // C-layout -> A-layout transpose of P via LDS
#pragma unroll
    for (int r = 0; r < 8; ++r) {
      T[(half * 8 + r) * 32 + l16] = f2bf(p0a[r]);
      T[(half * 8 + r) * 32 + 16 + l16] = f2bf(p1a[r]);
    }
    __syncthreads();
    Frag32 pf;
    pf.lo = *(const uint4*)&T[l16 * 32 + half * 16];
    pf.hi = *(const uint4*)&T[l16 * 32 + 8 + half * 16];
    v16bf pa = as_v16bf(pf);
    __syncthreads();
#pragma unroll
    for (int dd = 0; dd < 4; ++dd) {
      const unsigned short* vr =
          vT + ((size_t)(h * HD + dd * 16 + l16)) * NTOK + off0 + k0 + half * 16;
      Frag32 vf; vf.lo = *(const uint4*)vr; vf.hi = *(const uint4*)(vr + 8);
      o[dd] = wmma_bf16(pa, as_v16bf(vf), o[dd]);
    }
  }
#pragma unroll
  for (int r = 0; r < 8; ++r) {
    int q = q0 + half * 8 + r;
    if (q < len) {
      float inv = 1.0f / L[r];
#pragma unroll
      for (int dd = 0; dd < 4; ++dd)
        ob[((size_t)(off0 + q)) * DM + h * HD + dd * 16 + l16] = f2bf(o[dd][r] * inv);
    }
  }
}

// ---------------------------------------------------------------------------
// Out projection + residual: out[N,1024] = ob @ w_out + x (f32 output)
// ---------------------------------------------------------------------------
__global__ void __launch_bounds__(32) gemm_out_kernel(
    const unsigned short* __restrict__ ob, const unsigned short* __restrict__ wT,
    const float* __restrict__ x, float* __restrict__ out) {
  const int MT = NTOK / 16;
  int wid = blockIdx.x;
  int mt = wid % MT, nt = wid / MT;         // nt: 0..15
  int lane = threadIdx.x, half = lane >> 4, l16 = lane & 15;
  int row0 = mt * 16, col0 = nt * 64;
  v8f acc[4] = {};
  const unsigned short* arow = ob + (size_t)(row0 + l16) * DM;
  for (int k0 = 0; k0 < DM; k0 += 32) {
    Frag32 af;
    af.lo = *(const uint4*)(arow + k0 + half * 16);
    af.hi = *(const uint4*)(arow + k0 + 8 + half * 16);
    v16bf av = as_v16bf(af);
#pragma unroll
    for (int c = 0; c < 4; ++c) {
      const unsigned short* brow = wT + (size_t)(col0 + c * 16 + l16) * DM + k0 + half * 16;
      Frag32 bf; bf.lo = *(const uint4*)brow; bf.hi = *(const uint4*)(brow + 8);
      acc[c] = wmma_bf16(av, as_v16bf(bf), acc[c]);
    }
  }
#pragma unroll
  for (int c = 0; c < 4; ++c) {
#pragma unroll
    for (int r = 0; r < 8; ++r) {
      size_t row = row0 + half * 8 + r;
      size_t col = col0 + c * 16 + l16;
      out[row * DM + col] = acc[c][r] + x[row * DM + col];
    }
  }
}

// ---------------------------------------------------------------------------
extern "C" void kernel_launch(void* const* d_in, const int* in_sizes, int n_in,
                              void* d_out, int out_size, void* d_ws, size_t ws_size,
                              hipStream_t stream) {
  (void)in_sizes; (void)n_in; (void)out_size; (void)ws_size;
  const float* x     = (const float*)d_in[0];
  const float* pos   = (const float*)d_in[1];
  const int*   boff  = (const int*)d_in[2];
  const float* gamma = (const float*)d_in[3];
  const float* beta  = (const float*)d_in[4];
  const float* wqkv  = (const float*)d_in[5];
  const float* wout  = (const float*)d_in[6];
  const float* freqs = (const float*)d_in[7];
  float* out = (float*)d_out;

  char* p = (char*)d_ws;
  auto carve = [&](size_t bytes) { char* r = p; p += (bytes + 255) & ~(size_t)255; return r; };
  unsigned short* wqkvT = (unsigned short*)carve((size_t)3 * DM * DM * 2);   // [3072,1024] bf16
  unsigned short* woutT = (unsigned short*)carve((size_t)DM * DM * 2);       // [1024,1024] bf16
  unsigned short* hbuf  = (unsigned short*)carve((size_t)NTOK * DM * 2);     // LN output bf16
  unsigned short* qb    = (unsigned short*)carve((size_t)NH * NTOK * HD * 2);
  unsigned short* kb    = (unsigned short*)carve((size_t)NH * NTOK * HD * 2);
  unsigned short* vTb   = (unsigned short*)carve((size_t)NH * NTOK * HD * 2);
  unsigned short* ob    = hbuf;  // reuse: hbuf dead after QKV GEMM

  convT_kernel<<<(DM * 3 * DM + 255) / 256, 256, 0, stream>>>(wqkv, wqkvT, DM, 3 * DM);
  convT_kernel<<<(DM * DM + 255) / 256, 256, 0, stream>>>(wout, woutT, DM, DM);
  ln_kernel<<<NTOK, 256, 0, stream>>>(x, gamma, beta, hbuf);
  gemm_qkv_kernel<<<(NTOK / 16) * (3 * DM / 64), 32, 0, stream>>>(hbuf, wqkvT, qb, kb, vTb);
  rope_kernel<<<(NTOK * NH * 32 + 255) / 256, 256, 0, stream>>>(pos, freqs, qb, kb);
  attn_kernel<<<NBAT * NH * 64, 32, 0, stream>>>(boff, qb, kb, vTb, ob);
  gemm_out_kernel<<<(NTOK / 16) * (DM / 64), 32, 0, stream>>>(ob, woutT, x, out);
}
